// SimpleBlock_19825569038989
// MI455X (gfx1250) — compile-verified, matrix operand order checked by reading, not run on hardware
//
#include <hip/hip_runtime.h>

#define N_TOKENS 524288
#define DIM 128
#define HIDDEN 256
#define ROWS_PER_WG 256

typedef __attribute__((ext_vector_type(16))) __bf16        v16bf;
typedef __attribute__((ext_vector_type(8)))  float         v8f;
typedef __attribute__((ext_vector_type(4)))  float         f32x4;
typedef __attribute__((ext_vector_type(4)))  unsigned int  u32x4;

union BfPack {
    v16bf        bf;
    unsigned int w[8];
    u32x4        q[2];
};

__device__ __forceinline__ unsigned short f32_to_bf16_rn(float f) {
    unsigned int u = __float_as_uint(f);
    unsigned int r = u + 0x7FFFu + ((u >> 16) & 1u);   // round-to-nearest-even
    return (unsigned short)(r >> 16);
}
__device__ __forceinline__ float bf16_us_to_f32(unsigned short h) {
    return __uint_as_float(((unsigned int)h) << 16);
}

// Pack the high 16 bits (bf16 truncation) of two f32 bit patterns into one
// dword with a single v_perm_b32: result = { b1[31:16], b0[31:16] }.
__device__ __forceinline__ unsigned int pack_bf16_trunc(unsigned int b1, unsigned int b0) {
    return __builtin_amdgcn_perm(b1, b0, 0x07060302u);
}

// ---------------------------------------------------------------------------
// Kernel 1: fold the two back-to-back weight matrices into one 128x128 matrix
//   M[i][j] = sum_k (w_proj[k][i] + w_untouched[k][i]) * w_experts[0][j][k]
// and store bf16 hi/lo splits TRANSPOSED (M_T[j][i]) for contiguous B gathers.
// ---------------------------------------------------------------------------
__global__ void fold_weights_kernel(const float* __restrict__ w_proj,
                                    const float* __restrict__ w_unt,
                                    const float* __restrict__ w_exp,
                                    unsigned short* __restrict__ m_hiT,
                                    unsigned short* __restrict__ m_loT) {
    int idx = blockIdx.x * blockDim.x + threadIdx.x;   // 0 .. 16383
    int j = idx >> 7;        // output column (0..127)
    int i = idx & 127;       // contraction index with x (0..127)
    float acc = 0.f;
    #pragma unroll 4
    for (int k = 0; k < HIDDEN; ++k) {
        float w1 = w_proj[k * DIM + i] + w_unt[k * DIM + i];
        acc += w1 * w_exp[j * HIDDEN + k];   // expert group 0
    }
    unsigned short hi = f32_to_bf16_rn(acc);
    unsigned short lo = f32_to_bf16_rn(acc - bf16_us_to_f32(hi));
    m_hiT[j * DIM + i] = hi;
    m_loT[j * DIM + i] = lo;
}

// ---------------------------------------------------------------------------
// Kernel 2: out(N,128) = x(N,128) @ M(128,128) via bf16 split WMMA.
// 8 waves / WG; wave w owns output column-tile w and holds its B operands
// (4 k-tiles x {hi,lo}) in registers for the whole kernel.
// ---------------------------------------------------------------------------
__global__ __launch_bounds__(256)
void moe_gemm_kernel(const float* __restrict__ x,
                     const unsigned short* __restrict__ m_hiT,
                     const unsigned short* __restrict__ m_loT,
                     float* __restrict__ out) {
    const int lane = threadIdx.x & 31;
    const int wave = threadIdx.x >> 5;      // n-tile 0..7
    const int half = lane >> 4;             // 0 or 1 (K-interleave half)
    const int lrow = lane & 15;             // row (A) / col (B,D) within tile
    const int kofs = half * 8;

    // ---- B operands: M_T[col][k], per the 16-bit B-matrix VGPR layout ----
    // element h of v16bf: k = 32*kt + (h<8 ? 0 : 16) + (lane>>4)*8 + (h&7)
    const int ncol = wave * 16 + lrow;
    BfPack bhi[4], blo[4];
    #pragma unroll
    for (int kt = 0; kt < 4; ++kt) {
        int kb = kt * 32 + kofs;
        bhi[kt].q[0] = *(const u32x4*)(m_hiT + ncol * DIM + kb);
        bhi[kt].q[1] = *(const u32x4*)(m_hiT + ncol * DIM + kb + 16);
        blo[kt].q[0] = *(const u32x4*)(m_loT + ncol * DIM + kb);
        blo[kt].q[1] = *(const u32x4*)(m_loT + ncol * DIM + kb + 16);
    }

    const long row_base = (long)blockIdx.x * ROWS_PER_WG;

    for (int rg = 0; rg < ROWS_PER_WG / 16; ++rg) {
        const long  arow = row_base + rg * 16 + lrow;      // A-matrix row
        const float* xr  = x + arow * DIM;

        // prefetch this lane's row for the next 16-row group
        __builtin_prefetch(xr + 16 * DIM, 0, 0);

        v8f acc = {0.f, 0.f, 0.f, 0.f, 0.f, 0.f, 0.f, 0.f};

        #pragma unroll
        for (int kt = 0; kt < 4; ++kt) {
            int kb = kt * 32 + kofs;
            u32x4 f0 = *(const u32x4*)(xr + kb);        // raw f32 bit patterns
            u32x4 f1 = *(const u32x4*)(xr + kb + 4);
            u32x4 f2 = *(const u32x4*)(xr + kb + 16);
            u32x4 f3 = *(const u32x4*)(xr + kb + 20);

            unsigned int fb[16];
            #pragma unroll
            for (int e = 0; e < 4; ++e) {
                fb[e]      = f0[e];
                fb[4 + e]  = f1[e];
                fb[8 + e]  = f2[e];
                fb[12 + e] = f3[e];
            }

            // Truncation split: hi = f & 0xffff0000 (exact bf16),
            // lo = f - hi (exact), lo_bf16 = truncate(lo). ~2^-16 rel error.
            BfPack ahi, alo;
            #pragma unroll
            for (int p = 0; p < 8; ++p) {
                unsigned int b0 = fb[2 * p];
                unsigned int b1 = fb[2 * p + 1];
                ahi.w[p] = pack_bf16_trunc(b1, b0);
                float l0 = __uint_as_float(b0) - __uint_as_float(b0 & 0xffff0000u);
                float l1 = __uint_as_float(b1) - __uint_as_float(b1 & 0xffff0000u);
                alo.w[p] = pack_bf16_trunc(__float_as_uint(l1), __float_as_uint(l0));
            }

            // split-precision product: hi*hi + hi*lo + lo*hi  (~f32 accuracy)
            acc = __builtin_amdgcn_wmma_f32_16x16x32_bf16(
                      false, ahi.bf, false, bhi[kt].bf, (short)0, acc, false, false);
            acc = __builtin_amdgcn_wmma_f32_16x16x32_bf16(
                      false, ahi.bf, false, blo[kt].bf, (short)0, acc, false, false);
            acc = __builtin_amdgcn_wmma_f32_16x16x32_bf16(
                      false, alo.bf, false, bhi[kt].bf, (short)0, acc, false, false);
        }

        // D layout: VGPR r, lane l -> row = r + 8*(l>>4), col = l&15
        const long orow0 = row_base + rg * 16;
        const int  col   = wave * 16 + lrow;
        #pragma unroll
        for (int r = 0; r < 8; ++r) {
            // streamed once, no reuse -> non-temporal store
            __builtin_nontemporal_store(acc[r], &out[(orow0 + r + half * 8) * DIM + col]);
        }
    }
}

extern "C" void kernel_launch(void* const* d_in, const int* in_sizes, int n_in,
                              void* d_out, int out_size, void* d_ws, size_t ws_size,
                              hipStream_t stream) {
    const float* x      = (const float*)d_in[0];
    const float* w_proj = (const float*)d_in[1];
    const float* w_unt  = (const float*)d_in[2];
    const float* w_exp  = (const float*)d_in[3];
    float*       out    = (float*)d_out;

    unsigned short* m_hiT = (unsigned short*)d_ws;
    unsigned short* m_loT = m_hiT + DIM * DIM;

    fold_weights_kernel<<<(DIM * DIM) / 256, 256, 0, stream>>>(
        w_proj, w_unt, w_exp, m_hiT, m_loT);

    moe_gemm_kernel<<<N_TOKENS / ROWS_PER_WG, 256, 0, stream>>>(
        x, m_hiT, m_loT, out);
}
